// DeformableTransformerDecoderLayer_71433896067575
// MI455X (gfx1250) — matrix operand order repaired and checked by reading, not compiled, not run on hardware
//
#include <hip/hip_runtime.h>
#include <hip/hip_bf16.h>
#include <math.h>

// Problem constants (fixed by the reference)
#define DD   256
#define NHH  8
#define NLL  4
#define NPP  8
#define NQQ  300
#define BSS  8
#define DHH  32
#define SS   13294
#define NQP  320          // NQ padded to a multiple of 32 for WMMA K-loops
#define TT   (NQQ*BSS)    // 2400 tokens

typedef __attribute__((ext_vector_type(16))) _Float16 v16h;
typedef __attribute__((ext_vector_type(8)))  float    v8f;

__device__ __forceinline__ float wredsum(float v) {
#pragma unroll
  for (int o = 16; o > 0; o >>= 1) v += __shfl_xor(v, o, 32);
  return v;
}
__device__ __forceinline__ float wredmax(float v) {
#pragma unroll
  for (int o = 16; o > 0; o >>= 1) v = fmaxf(v, __shfl_xor(v, o, 32));
  return v;
}

// ---------------------------------------------------------------------------
// WMMA GEMM, NT 16x16 tiles per wave (16 x 16*NT output per wave):
//   C[m,n] = sum_k A[m,k] * B(k,n) (+bias[n]) (+ReLU)
//   A row-major (lda).  amode=1: A row m maps to memory[(m%S)*BS + m/S] (b,s remap).
//   b_nk=1: B is (N,K) row-major (weights, "X @ W^T").  b_nk=0: B is (K,N) row-major.
//   One A fragment per 32-K chunk feeds NT back-to-back v_wmma_f32_16x16x32_f16.
// ---------------------------------------------------------------------------
template <int NT>
__global__ __launch_bounds__(32) void wmma_gemm_t(
    const float* __restrict__ A, const float* __restrict__ Bw,
    const float* __restrict__ bias, float* __restrict__ C,
    int K, int lda, int ldb, int ldc,
    long sA, long sB, long sC,
    int amode, int b_nk, int relu)
{
  const int lane = threadIdx.x;
  const int m0 = blockIdx.x * 16, n0 = blockIdx.y * (16 * NT);
  const float* Ab = A  + (long)blockIdx.z * sA;
  const float* Bb = Bw + (long)blockIdx.z * sB;
  float*       Cb = C  + (long)blockIdx.z * sC;
  const int l15 = lane & 15, g = lane >> 4;
  const int mrow = m0 + l15;
  int aoff;                                 // all A offsets fit in 32 bits here
  if (amode) aoff = ((mrow % SS) * BSS + (mrow / SS)) * lda;   // (b,s) remap
  else       aoff = mrow * lda;
  const float* Ar = Ab + aoff;

  v8f acc[NT] = {};
  for (int kk = 0; kk < K; kk += 32) {
    // A fragment: 16x32 f16; lane-group g covers K = kk+16g..kk+16g+15
    v16h af;
    const float* ap = Ar + kk + g * 16;
#pragma unroll
    for (int j = 0; j < 8; ++j)  af[j] = (_Float16)ap[j];
#pragma unroll
    for (int j = 8; j < 16; ++j) af[j] = (_Float16)ap[j];
    // B fragments: 32x16 f16; lane-group g covers K = kk+16g..kk+16g+15
    const int kb = kk + g * 16;
#pragma unroll
    for (int t = 0; t < NT; ++t) {
      const int n = n0 + t * 16 + l15;
      v16h bf;
      if (b_nk) {
        const float* bp = Bb + n * ldb + kb;
#pragma unroll
        for (int j = 0; j < 16; ++j) bf[j] = (_Float16)bp[j];
      } else {
        const float* bp = Bb + kb * ldb + n;
#pragma unroll
        for (int j = 0; j < 16; ++j) bf[j] = (_Float16)bp[j * ldb];
      }
      acc[t] = __builtin_amdgcn_wmma_f32_16x16x32_f16(false, af, false, bf,
                                                      (short)0, acc[t], false, false);
    }
  }
#pragma unroll
  for (int t = 0; t < NT; ++t) {
    const int n = n0 + t * 16 + l15;
    const float bv = bias ? bias[n] : 0.0f;
#pragma unroll
    for (int r = 0; r < 8; ++r) {          // C layout: VGPR r -> row m0 + r + 8*g
      const int m = m0 + r + 8 * g;
      float v = acc[t][r] + bv;
      if (relu) v = fmaxf(v, 0.0f);
      Cb[(long)m * ldc + n] = v;
    }
  }
}

// ---------------------------------------------------------------------------
// Elementwise / reduction kernels
// ---------------------------------------------------------------------------
__global__ void ew_add(const float* __restrict__ a, const float* __restrict__ b,
                       float* __restrict__ o, int n) {
  int i = blockIdx.x * 256 + threadIdx.x;
  if (i < n) o[i] = a[i] + b[i];
}

// qk_out (T,512) & v_out (T,256) [token t = q*BS+b] -> padded (b,h,NQP,32) frags
__global__ void scatter_qkv(const float* __restrict__ qk, const float* __restrict__ vv,
                            float* __restrict__ qh, float* __restrict__ kh,
                            float* __restrict__ vh) {
  int i = blockIdx.x * 256 + threadIdx.x;
  const int total = 64 * NQP * DHH;
  if (i >= total) return;
  int d  = i & 31;
  int q  = (i >> 5) % NQP;
  int bh = i / (NQP * DHH);
  int b = bh >> 3, h = bh & 7;
  float a = 0.f, c = 0.f, e = 0.f;
  if (q < NQQ) {
    int t = q * BSS + b;
    a = qk[t * 512 + h * 32 + d];
    c = qk[t * 512 + 256 + h * 32 + d];
    e = vv[t * 256 + h * 32 + d];
  }
  qh[i] = a; kh[i] = c; vh[i] = e;
}

// Row softmax over 300 real columns of (64, NQP, NQP) logits; pads -> 0.
__global__ __launch_bounds__(32) void softmax_attn(float* __restrict__ L) {
  int bh = blockIdx.x, row = blockIdx.y, lane = threadIdx.x;
  float* Lr = L + ((long)bh * NQP + row) * NQP;
  const float scale = 0.17677669529663687f;   // 1/sqrt(32)
  float v[10]; float mx = -1e30f;
#pragma unroll
  for (int i = 0; i < 10; ++i) {
    int c = lane + i * 32;
    float x = (c < NQQ) ? Lr[c] * scale : -1e30f;
    v[i] = x; mx = fmaxf(mx, x);
  }
  mx = wredmax(mx);
  float s = 0.f;
#pragma unroll
  for (int i = 0; i < 10; ++i) {
    int c = lane + i * 32;
    float e = (c < NQQ) ? __expf(v[i] - mx) : 0.0f;
    v[i] = e; s += e;
  }
  s = wredsum(s);
  float inv = 1.0f / s;
#pragma unroll
  for (int i = 0; i < 10; ++i) Lr[lane + i * 32] = v[i] * inv;
}

// sa_p (b,h,NQP,32) -> sa_flat (t = q*BS+b, 256)
__global__ void gather_sa(const float* __restrict__ sp, float* __restrict__ sf) {
  int i = blockIdx.x * 256 + threadIdx.x;
  if (i >= TT * DD) return;
  int c = i & 255; int t = i >> 8;
  int q = t / BSS, b = t % BSS;
  int h = c >> 5, d = c & 31;
  sf[i] = sp[((b * 8 + h) * NQP + q) * DHH + d];
}

// out[row] = LayerNorm(x[row] + y[yrow]); remap=1: yrow is b-major (b*NQ+q)
__global__ __launch_bounds__(32) void add_ln(
    const float* __restrict__ x, const float* __restrict__ y,
    const float* __restrict__ gam, const float* __restrict__ bet,
    float* __restrict__ out, int remap) {
  int row = blockIdx.x, lane = threadIdx.x;
  int yrow = row;
  if (remap) { int q = row / BSS, b = row % BSS; yrow = b * NQQ + q; }
  const float* xr = x + (long)row * DD;
  const float* yr = y + (long)yrow * DD;
  float v[8]; float s = 0.f;
#pragma unroll
  for (int i = 0; i < 8; ++i) { int c = lane + i * 32; float t = xr[c] + yr[c]; v[i] = t; s += t; }
  s = wredsum(s);
  float mu = s * (1.0f / DD);
  float vs = 0.f;
#pragma unroll
  for (int i = 0; i < 8; ++i) { float d = v[i] - mu; vs += d * d; }
  vs = wredsum(vs);
  float inv = rsqrtf(vs * (1.0f / DD) + 1e-5f);
  float* orow = out + (long)row * DD;
#pragma unroll
  for (int i = 0; i < 8; ++i) {
    int c = lane + i * 32;
    orow[c] = (v[i] - mu) * inv * gam[c] + bet[c];
  }
}

// q2[(b*NQ+q)] = t1[(q*BS+b)] + mask[(q*BS+b)]
__global__ void make_q2(const float* __restrict__ t1, const float* __restrict__ msk,
                        float* __restrict__ q2) {
  int i = blockIdx.x * 256 + threadIdx.x;
  if (i >= TT * DD) return;
  int c = i & 255; int t = i >> 8;
  int q = t / BSS, b = t % BSS;
  q2[(b * NQQ + q) * DD + c] = t1[i] + msk[i];
}

// In-place softmax over the 32 (l,p) weights of each (b,q,h)
__global__ __launch_bounds__(32) void softmax_aw(float* __restrict__ A) {
  int r = blockIdx.x, h = blockIdx.y, lane = threadIdx.x;
  float* p = A + (long)r * DD + h * 32;
  float x = p[lane];
  float mx = wredmax(x);
  float e = __expf(x - mx);
  float s = wredsum(e);
  p[lane] = e / s;
}

// Multi-scale deformable sampling: one wave per (b*NQ+q, h); lane = channel d.
// (memory_key_padding_mask is all-false in this problem, so value masking is a no-op.)
__global__ __launch_bounds__(32) void msdeform(
    const float* __restrict__ value, const float* __restrict__ off,
    const float* __restrict__ aw, const float* __restrict__ refb,
    const int* __restrict__ spatial, const int* __restrict__ lstart,
    float* __restrict__ out) {
  int r = blockIdx.x, h = blockIdx.y, lane = threadIdx.x;
  int b = r / NQQ, q = r % NQQ;
  const float* rf = refb + (q * BSS + b) * 4;   // bboxs are (NQ,BS,4)
  float rx = rf[0], ry = rf[1], rw = rf[2], rh = rf[3];
  const float* vb   = value + ((long)b * SS) * DD + h * DHH + lane;
  const float* offr = off + (long)r * (NHH * NLL * NPP * 2);
  const float* awr  = aw  + (long)r * DD;
  float acc = 0.0f;
#pragma unroll
  for (int l = 0; l < NLL; ++l) {
    int H = spatial[l * 2 + 0], W = spatial[l * 2 + 1];
    int ls = lstart[l];
    float Hf = (float)H, Wf = (float)W;
    for (int p = 0; p < NPP; ++p) {
      int pi = (h * NLL + l) * NPP + p;
      float ox = offr[pi * 2 + 0], oy = offr[pi * 2 + 1];
      float a  = awr[pi];
      float lx = rx + ox * (0.5f / NPP) * rw;
      float ly = ry + oy * (0.5f / NPP) * rh;
      float xx = lx * Wf - 0.5f;
      float yy = ly * Hf - 0.5f;
      float x0 = floorf(xx), y0 = floorf(yy);
#pragma unroll
      for (int dx = 0; dx < 2; ++dx) {
#pragma unroll
        for (int dy = 0; dy < 2; ++dy) {
          float xi = x0 + dx, yi = y0 + dy;
          if (xi >= 0.0f && xi < Wf && yi >= 0.0f && yi < Hf) {
            float w = (1.0f - fabsf(xx - xi)) * (1.0f - fabsf(yy - yi));
            int idx = ls + (int)yi * W + (int)xi;
            acc += w * a * vb[(long)idx * DD];
          }
        }
      }
    }
  }
  out[(long)r * DD + h * DHH + lane] = acc;
}

// ---------------------------------------------------------------------------
extern "C" void kernel_launch(void* const* d_in, const int* in_sizes, int n_in,
                              void* d_out, int out_size, void* d_ws, size_t ws_size,
                              hipStream_t stream) {
  (void)in_sizes; (void)n_in; (void)out_size; (void)ws_size;
  const float* tgt    = (const float*)d_in[0];
  const float* qmask  = (const float*)d_in[1];
  const float* refb   = (const float*)d_in[2];
  /* d_in[3] tgt_reference_masks: unused by reference */
  const float* memory = (const float*)d_in[4];
  const float* inw    = (const float*)d_in[5];
  const float* inb    = (const float*)d_in[6];
  const float* osw    = (const float*)d_in[7];
  const float* osb    = (const float*)d_in[8];
  const float* n2g    = (const float*)d_in[9];
  const float* n2b    = (const float*)d_in[10];
  const float* vw     = (const float*)d_in[11];
  const float* vbias  = (const float*)d_in[12];
  const float* ofw    = (const float*)d_in[13];
  const float* ofb    = (const float*)d_in[14];
  const float* aww    = (const float*)d_in[15];
  const float* awb    = (const float*)d_in[16];
  const float* ocw    = (const float*)d_in[17];
  const float* ocb    = (const float*)d_in[18];
  const float* n1g    = (const float*)d_in[19];
  const float* n1b    = (const float*)d_in[20];
  const float* l1w    = (const float*)d_in[21];
  const float* l1b    = (const float*)d_in[22];
  const float* l2w    = (const float*)d_in[23];
  const float* l2b    = (const float*)d_in[24];
  const float* n3g    = (const float*)d_in[25];
  const float* n3b    = (const float*)d_in[26];
  /* d_in[27] memory_key_padding_mask: all-false, unused */
  const int* spatial  = (const int*)d_in[28];
  const int* lstart   = (const int*)d_in[29];
  float* out = (float*)d_out;

  // Workspace layout (floats) — dead buffers aliased for later stages.
  float* ws     = (float*)d_ws;
  float* qbuf   = ws;                      // T*D       (aliased later: q2)
  float* qk_out = qbuf   + 614400;         // T*512     (aliased later: off_o)
  float* v_out  = qk_out + 1228800;        // T*D       (aliased later: aw_o)
  float* qh_p   = v_out  + 614400;         // 64*NQP*32
  float* kh_p   = qh_p   + 655360;
  float* vh_p   = kh_p   + 655360;
  float* logits = vh_p   + 655360;         // 64*NQP*NQP (aliased later: ffn_h)
  float* sa_p   = logits + 6553600;        // 64*NQP*32
  float* sa_f   = sa_p   + 655360;         // T*D
  float* sa_pr  = sa_f   + 614400;         // T*D
  float* t1     = sa_pr  + 614400;         // T*D
  float* value  = t1     + 614400;         // BS*S*D = 27226112
  float* ca_pre = value  + 27226112;       // T*D
  float* ca_pr  = ca_pre + 614400;         // T*D
  float* t2     = ca_pr  + 614400;         // T*D
  float* ffn_o  = t2     + 614400;         // T*D
  float* q2    = qbuf;
  float* off_o = qk_out;
  float* aw_o  = v_out;
  float* ffn_h = logits;

  dim3 b32(32), b256(256);
  const int EW = (TT * DD + 255) / 256;

  // 1) q = tgt + query_mask
  ew_add<<<EW, b256, 0, stream>>>(tgt, qmask, qbuf, TT * DD);
  // 2) [q|k] projection: (T,256) @ in_proj_w[0:512]^T
  wmma_gemm_t<4><<<dim3(TT/16, 512/64, 1), b32, 0, stream>>>(qbuf, inw, inb, qk_out,
      256, 256, 256, 512, 0, 0, 0, 0, 1, 0);
  // 3) v projection from tgt
  wmma_gemm_t<4><<<dim3(TT/16, 256/64, 1), b32, 0, stream>>>(tgt, inw + 512*256, inb + 512, v_out,
      256, 256, 256, 256, 0, 0, 0, 0, 1, 0);
  // 4) scatter into padded (b,h,NQP,32) head fragments (zeros in pad rows)
  scatter_qkv<<<(64*NQP*32 + 255)/256, b256, 0, stream>>>(qk_out, v_out, qh_p, kh_p, vh_p);
  // 5) logits[b,h] = qh @ kh^T  (batched 64, K=32)
  wmma_gemm_t<4><<<dim3(NQP/16, NQP/64, 64), b32, 0, stream>>>(qh_p, kh_p, nullptr, logits,
      32, 32, 32, NQP, (long)NQP*32, (long)NQP*32, (long)NQP*NQP, 0, 1, 0);
  // 6) softmax (scale 1/sqrt(32), pad columns -> 0)
  softmax_attn<<<dim3(64, NQP), b32, 0, stream>>>(logits);
  // 7) sa[b,h] = attn @ vh  (batched 64, K=NQP, N=32 -> 2 tiles/wave)
  wmma_gemm_t<2><<<dim3(NQP/16, 1, 64), b32, 0, stream>>>(logits, vh_p, nullptr, sa_p,
      NQP, NQP, 32, 32, (long)NQP*NQP, (long)NQP*32, (long)NQP*32, 0, 0, 0);
  // 8) gather heads back to (t=q*BS+b, 256)
  gather_sa<<<EW, b256, 0, stream>>>(sa_p, sa_f);
  // 9) self-attn output projection
  wmma_gemm_t<4><<<dim3(TT/16, 256/64, 1), b32, 0, stream>>>(sa_f, osw, osb, sa_pr,
      256, 256, 256, 256, 0, 0, 0, 0, 1, 0);
  // 10) t1 = LN(tgt + sa)
  add_ln<<<TT, b32, 0, stream>>>(tgt, sa_pr, n2g, n2b, t1, 0);
  // 11) q2[(b,q)] = t1[(q,b)] + mask[(q,b)]
  make_q2<<<EW, b256, 0, stream>>>(t1, qmask, q2);
  // 12) sampling offsets (T,512)
  wmma_gemm_t<4><<<dim3(TT/16, 512/64, 1), b32, 0, stream>>>(q2, ofw, ofb, off_o,
      256, 256, 256, 512, 0, 0, 0, 0, 1, 0);
  // 13) attention weights (T,256), then per-head softmax over 32
  wmma_gemm_t<4><<<dim3(TT/16, 256/64, 1), b32, 0, stream>>>(q2, aww, awb, aw_o,
      256, 256, 256, 256, 0, 0, 0, 0, 1, 0);
  softmax_aw<<<dim3(TT, 8), b32, 0, stream>>>(aw_o);
  // 15) value projection over all memory tokens (amode=1: (b,s) remap of memory)
  wmma_gemm_t<4><<<dim3((BSS*SS)/16, 256/64, 1), b32, 0, stream>>>(memory, vw, vbias, value,
      256, 256, 256, 256, 0, 0, 0, 1, 1, 0);
  // 16) bilinear deformable sampling
  msdeform<<<dim3(TT, 8), b32, 0, stream>>>(value, off_o, aw_o, refb, spatial, lstart, ca_pre);
  // 17) cross-attn output projection
  wmma_gemm_t<4><<<dim3(TT/16, 256/64, 1), b32, 0, stream>>>(ca_pre, ocw, ocb, ca_pr,
      256, 256, 256, 256, 0, 0, 0, 0, 1, 0);
  // 18) t2 = LN(t1 + ca)  (ca rows are b-major -> remap)
  add_ln<<<TT, b32, 0, stream>>>(t1, ca_pr, n1g, n1b, t2, 1);
  // 19) FFN lin1 + ReLU
  wmma_gemm_t<4><<<dim3(TT/16, 1024/64, 1), b32, 0, stream>>>(t2, l1w, l1b, ffn_h,
      256, 256, 256, 1024, 0, 0, 0, 0, 1, 1);
  // 20) FFN lin2
  wmma_gemm_t<4><<<dim3(TT/16, 256/64, 1), b32, 0, stream>>>(ffn_h, l2w, l2b, ffn_o,
      1024, 1024, 1024, 256, 0, 0, 0, 0, 1, 0);
  // 21) out = LN(t2 + ffn)
  add_ln<<<TT, b32, 0, stream>>>(t2, ffn_o, n3g, n3b, out, 0);
}
